// SEIR_LSTM_377957122348
// MI455X (gfx1250) — compile-verified, matrix operand order checked by reading, not run on hardware
//
#include <hip/hip_runtime.h>
#include <hip/hip_bf16.h>

typedef __attribute__((ext_vector_type(16))) _Float16 v16h;
typedef __attribute__((ext_vector_type(8)))  float    v8f;
typedef _Float16 h16;

#define BB  64
#define TT  256
#define DD  32
#define HH  512
#define NHH 8
#define HDD 64
#define MLL 512
#define BTT (BB*TT)

__device__ __forceinline__ float gelu_exact(float x){
  return 0.5f * x * (1.f + erff(x * 0.70710678118654752f));
}
__device__ __forceinline__ float sigmf(float x){
  return 1.f / (1.f + __expf(-x));
}

// ---------------------------------------------------------------------------
// WMMA fragment helpers (CDNA5 16x16x32 f16, wave32).  ISA 7.12.2 layouts:
//   A (16x32, MxK): lanes 0-15 hold M=lane, K={0..7,16..23}; lanes 16-31 hold
//                   M=lane-16, K={8..15,24..31}
//   B (32x16, KxN): mirrored, lane indexes column N
//   C/D (16x16):    VGPR i -> row (hs*8+i), lane&15 -> col
// ---------------------------------------------------------------------------
__device__ __forceinline__ v16h ld_frag_a(const h16* p, int lda){
  int lane = threadIdx.x & 31;
  int r = lane & 15, hs = lane >> 4;
  const h16* row = p + (size_t)r * (size_t)lda;
  v16h v;
  int k0 = hs * 8, k1 = 16 + hs * 8;
#pragma unroll
  for (int i = 0; i < 8; ++i) v[i]     = row[k0 + i];
#pragma unroll
  for (int i = 0; i < 8; ++i) v[8 + i] = row[k1 + i];
  return v;
}
// B(k,n) = p[k*ldk + n*ldn]
__device__ __forceinline__ v16h ld_frag_b(const h16* p, int ldk, int ldn){
  int lane = threadIdx.x & 31;
  int n = lane & 15, hs = lane >> 4;
  v16h v;
  int k0 = hs * 8, k1 = 16 + hs * 8;
#pragma unroll
  for (int i = 0; i < 8; ++i) v[i]     = p[(size_t)(k0 + i) * ldk + (size_t)n * ldn];
#pragma unroll
  for (int i = 0; i < 8; ++i) v[8 + i] = p[(size_t)(k1 + i) * ldk + (size_t)n * ldn];
  return v;
}
__device__ __forceinline__ v16h ld_frag_b_clampk(const h16* p, int ldk, int ldn, int kmax){
  int lane = threadIdx.x & 31;
  int n = lane & 15, hs = lane >> 4;
  v16h v;
  int k0 = hs * 8, k1 = 16 + hs * 8;
#pragma unroll
  for (int i = 0; i < 8; ++i){
    int ka = k0 + i; if (ka > kmax) ka = kmax;
    int kb = k1 + i; if (kb > kmax) kb = kmax;
    v[i]     = p[(size_t)ka * ldk + (size_t)n * ldn];
    v[8 + i] = p[(size_t)kb * ldk + (size_t)n * ldn];
  }
  return v;
}
__device__ __forceinline__ v8f wmma16(v16h a, v16h b, v8f c){
  return __builtin_amdgcn_wmma_f32_16x16x32_f16(false, a, false, b, (short)0, c, false, false);
}

// ---------------------------------------------------------------------------
// Casting helpers
// ---------------------------------------------------------------------------
__global__ void k_cast_h16(const float* __restrict__ s, h16* __restrict__ d, long n){
  long i = (long)blockIdx.x * blockDim.x + threadIdx.x;
  if (i < n) d[i] = (h16)s[i];
}
// conv weight [512][512][3] -> 3 tap matrices [512*512] f16
__global__ void k_cast_convw(const float* __restrict__ w, h16* __restrict__ d){
  long i = (long)blockIdx.x * blockDim.x + threadIdx.x;
  if (i < (long)HH * HH * 3){
    long nc = i / 3; int tap = (int)(i % 3);
    d[(size_t)tap * HH * HH + nc] = (h16)w[i];
  }
}

// ---------------------------------------------------------------------------
// Input LayerNorm over D=32 (wave-per-row, lane-per-channel)
// ---------------------------------------------------------------------------
__global__ void k_ln32(const float* __restrict__ x, const float* __restrict__ w,
                       const float* __restrict__ b, h16* __restrict__ out){
  int row  = blockIdx.x * 8 + (threadIdx.x >> 5);
  int lane = threadIdx.x & 31;
  float v  = x[(size_t)row * DD + lane];
  float mu = v;
  for (int o = 16; o > 0; o >>= 1) mu += __shfl_xor(mu, o, 32);
  mu *= (1.f / 32.f);
  float d = v - mu, var = d * d;
  for (int o = 16; o > 0; o >>= 1) var += __shfl_xor(var, o, 32);
  var *= (1.f / 32.f);
  float y = d * rsqrtf(var + 1e-5f) * w[lane] + b[lane];
  out[(size_t)row * DD + lane] = (h16)y;
}

// ---------------------------------------------------------------------------
// Row LayerNorm over 512, optional fused (a + g*m) input; writes f32/f16
// ---------------------------------------------------------------------------
__global__ void k_ln512(const float* __restrict__ a, const float* __restrict__ gmul,
                        const float* __restrict__ bv, const float* __restrict__ w,
                        const float* __restrict__ b, float* __restrict__ outf,
                        h16* __restrict__ outh){
  int row = blockIdx.x, t = threadIdx.x;
  __shared__ float red[8];
  __shared__ float sh_mu, sh_rs;
  size_t base = (size_t)row * HH;
  float v0 = a[base + t], v1 = a[base + t + 256];
  if (gmul){
    v0 += gmul[base + t] * bv[base + t];
    v1 += gmul[base + t + 256] * bv[base + t + 256];
  }
  float s = v0 + v1;
  for (int o = 16; o > 0; o >>= 1) s += __shfl_xor(s, o, 32);
  if ((t & 31) == 0) red[t >> 5] = s;
  __syncthreads();
  if (t == 0){ float tot = 0; for (int i = 0; i < 8; ++i) tot += red[i]; sh_mu = tot * (1.f/512.f); }
  __syncthreads();
  float mu = sh_mu;
  float d0 = v0 - mu, d1 = v1 - mu;
  float q = d0 * d0 + d1 * d1;
  for (int o = 16; o > 0; o >>= 1) q += __shfl_xor(q, o, 32);
  if ((t & 31) == 0) red[t >> 5] = q;
  __syncthreads();
  if (t == 0){ float tot = 0; for (int i = 0; i < 8; ++i) tot += red[i]; sh_rs = rsqrtf(tot * (1.f/512.f) + 1e-5f); }
  __syncthreads();
  float rs = sh_rs;
  float y0 = d0 * rs * w[t]       + b[t];
  float y1 = d1 * rs * w[t + 256] + b[t + 256];
  if (outf){ outf[base + t] = y0; outf[base + t + 256] = y1; }
  if (outh){ outh[base + t] = (h16)y0; outh[base + t + 256] = (h16)y1; }
}

// ---------------------------------------------------------------------------
// Generic WMMA GEMM:  C[M,N] = act(A16[M,K] @ W16[N,K]^T + bias) (+res)
// Block tile 64x64, 8 waves: wave = (mt = w&3, nh = w>>2) -> two 16x16 tiles
// ---------------------------------------------------------------------------
template<int ACT, bool RES>
__global__ void k_gemm(const h16* __restrict__ A, const h16* __restrict__ W,
                       const float* __restrict__ bias, const float* __restrict__ res,
                       float* __restrict__ Cf, h16* __restrict__ Ch,
                       int M, int N, int K, int ldh){
  __shared__ h16 sA[64 * 32];
  int bm = blockIdx.y * 64, bn = blockIdx.x * 64;
  int wid = threadIdx.x >> 5;
  int mt = wid & 3, nh = wid >> 2;
  v8f acc0 = {0,0,0,0,0,0,0,0}, acc1 = acc0;
  for (int k0 = 0; k0 < K; k0 += 32){
    __syncthreads();
    for (int idx = threadIdx.x; idx < 64 * 32; idx += 256){
      int r = idx >> 5, c = idx & 31;
      sA[idx] = A[(size_t)(bm + r) * K + k0 + c];
    }
    __builtin_prefetch((const void*)(W + (size_t)(bn + nh * 32) * K + k0 + 32), 0, 3);
    __syncthreads();
    v16h fa  = ld_frag_a(sA + mt * 16 * 32, 32);
    v16h fb0 = ld_frag_b(W + (size_t)(bn + nh * 32)      * K + k0, 1, K);
    v16h fb1 = ld_frag_b(W + (size_t)(bn + nh * 32 + 16) * K + k0, 1, K);
    acc0 = wmma16(fa, fb0, acc0);
    acc1 = wmma16(fa, fb1, acc1);
  }
  int lane = threadIdx.x & 31, nc = lane & 15, hs = lane >> 4;
  int m0 = bm + mt * 16 + hs * 8;
#pragma unroll
  for (int pr = 0; pr < 2; ++pr){
    v8f& a = pr ? acc1 : acc0;
    int n = bn + nh * 32 + pr * 16 + nc;
#pragma unroll
    for (int i = 0; i < 8; ++i){
      int m = m0 + i;
      float v = a[i] + (bias ? bias[n] : 0.f);
      if (ACT == 1) v = gelu_exact(v);
      else if (ACT == 2) v = sigmf(v);
      if (RES) v += res[(size_t)m * N + n];
      if (Cf) Cf[(size_t)m * N + n] = v;
      if (Ch) Ch[(size_t)m * ldh + n] = (h16)v;
    }
  }
}

// ---------------------------------------------------------------------------
// Dilated causal conv (kernel 3) as 3 shifted GEMMs + fused BN/GELU/residual
// ---------------------------------------------------------------------------
__global__ void k_convgemm(const h16* __restrict__ X, const h16* __restrict__ W3,
                           const float* __restrict__ cb, const float* __restrict__ bng,
                           const float* __restrict__ bnb, const float* __restrict__ xres,
                           float* __restrict__ Cf, int dil){
  __shared__ h16 sA[3 * 64 * 32];
  int bm = blockIdx.y * 64, bn = blockIdx.x * 64;
  int bb = bm / TT, t0 = bm % TT;
  int wid = threadIdx.x >> 5;
  int mt = wid & 3, nh = wid >> 2;
  v8f acc0 = {0,0,0,0,0,0,0,0}, acc1 = acc0;
  for (int k0 = 0; k0 < HH; k0 += 32){
    __syncthreads();
    for (int idx = threadIdx.x; idx < 3 * 64 * 32; idx += 256){
      int tap = idx >> 11, r = (idx >> 5) & 63, c = idx & 31;
      int t = t0 + r - tap * dil;
      h16 v = (h16)0.f;
      if (t >= 0) v = X[((size_t)bb * TT + t) * HH + k0 + c];
      sA[idx] = v;
    }
    __syncthreads();
#pragma unroll
    for (int tap = 0; tap < 3; ++tap){
      v16h fa  = ld_frag_a(sA + tap * 2048 + mt * 16 * 32, 32);
      v16h fb0 = ld_frag_b(W3 + (size_t)tap * HH * HH + (size_t)(bn + nh * 32)      * HH + k0, 1, HH);
      v16h fb1 = ld_frag_b(W3 + (size_t)tap * HH * HH + (size_t)(bn + nh * 32 + 16) * HH + k0, 1, HH);
      acc0 = wmma16(fa, fb0, acc0);
      acc1 = wmma16(fa, fb1, acc1);
    }
  }
  const float bnscale = rsqrtf(1.f + 1e-5f);
  int lane = threadIdx.x & 31, nc = lane & 15, hs = lane >> 4;
  int m0 = bm + mt * 16 + hs * 8;
#pragma unroll
  for (int pr = 0; pr < 2; ++pr){
    v8f& a = pr ? acc1 : acc0;
    int n = bn + nh * 32 + pr * 16 + nc;
#pragma unroll
    for (int i = 0; i < 8; ++i){
      int m = m0 + i;
      float v = (a[i] + cb[n]) * bnscale * bng[n] + bnb[n];
      v = gelu_exact(v) + xres[(size_t)m * HH + n];
      Cf[(size_t)m * HH + n] = v;
    }
  }
}

// ---------------------------------------------------------------------------
// Temporal gate: softmax(linear(concat(c1,c2,c3))) mix  (wave per row)
// ---------------------------------------------------------------------------
__global__ void k_tmix(const float* __restrict__ c1, const float* __restrict__ c2,
                       const float* __restrict__ c3, const float* __restrict__ tw,
                       const float* __restrict__ tb, float* __restrict__ xf,
                       h16* __restrict__ xh){
  int row  = blockIdx.x * 8 + (threadIdx.x >> 5);
  int lane = threadIdx.x & 31;
  const float* rows[3] = { c1 + (size_t)row * HH, c2 + (size_t)row * HH, c3 + (size_t)row * HH };
  float l0 = 0, l1 = 0, l2 = 0;
  for (int seg = 0; seg < 3; ++seg)
    for (int c = lane; c < HH; c += 32){
      float v = rows[seg][c];
      l0 += v * tw[0 * 3 * HH + seg * HH + c];
      l1 += v * tw[1 * 3 * HH + seg * HH + c];
      l2 += v * tw[2 * 3 * HH + seg * HH + c];
    }
  for (int o = 16; o > 0; o >>= 1){
    l0 += __shfl_xor(l0, o, 32); l1 += __shfl_xor(l1, o, 32); l2 += __shfl_xor(l2, o, 32);
  }
  l0 += tb[0]; l1 += tb[1]; l2 += tb[2];
  float mx = fmaxf(l0, fmaxf(l1, l2));
  float e0 = __expf(l0 - mx), e1 = __expf(l1 - mx), e2 = __expf(l2 - mx);
  float inv = 1.f / (e0 + e1 + e2);
  float w0 = e0 * inv, w1 = e1 * inv, w2 = e2 * inv;
  for (int c = lane; c < HH; c += 32){
    float v = w0 * rows[0][c] + w1 * rows[1][c] + w2 * rows[2][c];
    xf[(size_t)row * HH + c] = v;
    xh[(size_t)row * HH + c] = (h16)v;
  }
}

// ---------------------------------------------------------------------------
// Persistent LSTM layer: 16 WGs (one cluster), WG owns 32 hidden units
// (128 gate rows).  h exchanged via L2 double-buffer; cluster barrier per step.
// ---------------------------------------------------------------------------
__global__ void k_lstm(const h16* __restrict__ Xseq, const h16* __restrict__ Wih,
                       const h16* __restrict__ Whh, const float* __restrict__ bih,
                       const float* __restrict__ bhh, h16* __restrict__ hbuf,
                       h16* __restrict__ outH, int ldo, h16* __restrict__ outH2,
                       int ldo2, float* __restrict__ outF){
  __shared__ float zbuf[64 * 128];
  __shared__ float cstate[64 * 32];
  int wg  = blockIdx.x;        // 0..15
  int wid = threadIdx.x >> 5;  // 0..7 -> n-tile
  int lane = threadIdx.x & 31, nc = lane & 15, hs = lane >> 4;
  int gate = wid >> 1, ch0 = (wid & 1) * 16;
  int grow0 = gate * HH + wg * 32 + ch0;

  for (int i = threadIdx.x; i < 64 * 32; i += 256){
    cstate[i] = 0.f;
    int b = i >> 5, ch = i & 31;
    hbuf[(size_t)b * HH + wg * 32 + ch] = (h16)0.f;       // zero ping buffer
  }
  __threadfence();
  __syncthreads();
  __builtin_amdgcn_s_cluster_barrier();

  for (int t = 0; t < TT; ++t){
    const h16* hread  = hbuf + (size_t)(t & 1) * BB * HH;
    h16*       hwrite = hbuf + (size_t)((t + 1) & 1) * BB * HH;
    v8f acc[4];
#pragma unroll
    for (int mt = 0; mt < 4; ++mt) acc[mt] = (v8f){0,0,0,0,0,0,0,0};
    // z += x_t @ Wih^T (slice)
    for (int k0 = 0; k0 < HH; k0 += 32){
      v16h fb = ld_frag_b(Wih + (size_t)grow0 * HH + k0, 1, HH);
#pragma unroll
      for (int mt = 0; mt < 4; ++mt){
        v16h fa = ld_frag_a(Xseq + ((size_t)(mt * 16) * TT + t) * HH + k0, TT * HH);
        acc[mt] = wmma16(fa, fb, acc[mt]);
      }
    }
    // z += h_{t-1} @ Whh^T (slice)
    for (int k0 = 0; k0 < HH; k0 += 32){
      v16h fb = ld_frag_b(Whh + (size_t)grow0 * HH + k0, 1, HH);
#pragma unroll
      for (int mt = 0; mt < 4; ++mt){
        v16h fa = ld_frag_a(hread + (size_t)(mt * 16) * HH + k0, HH);
        acc[mt] = wmma16(fa, fb, acc[mt]);
      }
    }
    __syncthreads();
#pragma unroll
    for (int mt = 0; mt < 4; ++mt)
#pragma unroll
      for (int i = 0; i < 8; ++i)
        zbuf[(size_t)(mt * 16 + hs * 8 + i) * 128 + wid * 16 + nc] = acc[mt][i];
    __syncthreads();
    // gate math for this WG's 32 hidden units, all 64 batch rows
    for (int i = threadIdx.x; i < 64 * 32; i += 256){
      int b = i >> 5, ch = i & 31;
      int u = wg * 32 + ch;
      float iv = zbuf[(size_t)b * 128 +  0 + ch] + bih[0 * HH + u] + bhh[0 * HH + u];
      float fv = zbuf[(size_t)b * 128 + 32 + ch] + bih[1 * HH + u] + bhh[1 * HH + u];
      float gv = zbuf[(size_t)b * 128 + 64 + ch] + bih[2 * HH + u] + bhh[2 * HH + u];
      float ov = zbuf[(size_t)b * 128 + 96 + ch] + bih[3 * HH + u] + bhh[3 * HH + u];
      float cnew = sigmf(fv) * cstate[i] + sigmf(iv) * tanhf(gv);
      float hnew = sigmf(ov) * tanhf(cnew);
      cstate[i] = cnew;
      hwrite[(size_t)b * HH + u] = (h16)hnew;
      size_t orow = (size_t)b * TT + t;
      outH[orow * ldo + u] = (h16)hnew;
      if (outH2) outH2[orow * ldo2 + u] = (h16)hnew;
      if (outF)  outF[orow * HH + u] = hnew;
    }
    __threadfence();
    __syncthreads();
    __builtin_amdgcn_s_cluster_barrier();
  }
}

// ---------------------------------------------------------------------------
// Flash attention: one wave per (q-tile, head, batch).  WMMA QK^T and P.V,
// online softmax on C-fragment layout, rel-pos bias + causal mask fused.
// ---------------------------------------------------------------------------
__global__ void k_attn(const h16* __restrict__ Q, const h16* __restrict__ Kt,
                       const h16* __restrict__ V, const float* __restrict__ relpos,
                       h16* __restrict__ AO){
  int qt = blockIdx.x, hh = blockIdx.y, bb = blockIdx.z;
  int lane = threadIdx.x & 31, nc = lane & 15, hs = lane >> 4;
  __shared__ h16 sP[16 * 32];
  const float scale = 0.125f;   // 1/sqrt(64)
  const h16* qbase = Q + ((size_t)bb * TT + qt * 16) * HH + hh * HDD;
  v16h qa0 = ld_frag_a(qbase, HH);
  v16h qa1 = ld_frag_a(qbase + 32, HH);
  v8f o[4];
#pragma unroll
  for (int i = 0; i < 4; ++i) o[i] = (v8f){0,0,0,0,0,0,0,0};
  float mrun[8], lrun[8];
#pragma unroll
  for (int i = 0; i < 8; ++i){ mrun[i] = -1e30f; lrun[i] = 0.f; }
  int qrow0 = qt * 16 + hs * 8;

  for (int s0 = 0; s0 <= qt * 16; s0 += 32){
    float sv[2][8];
#pragma unroll
    for (int sub = 0; sub < 2; ++sub){
      int st = s0 + sub * 16;
      bool active = (st <= qt * 16);
      v8f sacc = {0,0,0,0,0,0,0,0};
      if (active){
        const h16* kb = Kt + ((size_t)bb * TT + st) * HH + hh * HDD;
        sacc = wmma16(qa0, ld_frag_b(kb,      1, HH), sacc);
        sacc = wmma16(qa1, ld_frag_b(kb + 32, 1, HH), sacc);
      }
#pragma unroll
      for (int i = 0; i < 8; ++i){
        int qtime = qrow0 + i, ktime = st + nc;
        float v = -1e30f;
        if (active && ktime <= qtime)
          v = sacc[i] * scale + relpos[((size_t)hh * MLL + qtime) * MLL + ktime];
        sv[sub][i] = v;
      }
    }
    float mnew[8], resc[8];
#pragma unroll
    for (int i = 0; i < 8; ++i){
      float m = fmaxf(sv[0][i], sv[1][i]);
      for (int od = 1; od < 16; od <<= 1) m = fmaxf(m, __shfl_xor(m, od, 32));
      mnew[i] = fmaxf(mrun[i], m);
      resc[i] = __expf(mrun[i] - mnew[i]);
    }
#pragma unroll
    for (int i = 0; i < 8; ++i){
      float p0 = __expf(sv[0][i] - mnew[i]);
      float p1 = __expf(sv[1][i] - mnew[i]);
      float rs = p0 + p1;
      for (int od = 1; od < 16; od <<= 1) rs += __shfl_xor(rs, od, 32);
      lrun[i] = lrun[i] * resc[i] + rs;
      mrun[i] = mnew[i];
      sP[(hs * 8 + i) * 32 + nc]      = (h16)p0;
      sP[(hs * 8 + i) * 32 + 16 + nc] = (h16)p1;
      o[0][i] *= resc[i]; o[1][i] *= resc[i]; o[2][i] *= resc[i]; o[3][i] *= resc[i];
    }
    __syncthreads();
    v16h pa = ld_frag_a(sP, 32);
    const h16* vb = V + ((size_t)bb * TT + s0) * HH + hh * HDD;
    int kmax = TT - 1 - s0;
#pragma unroll
    for (int nt = 0; nt < 4; ++nt)
      o[nt] = wmma16(pa, ld_frag_b_clampk(vb + nt * 16, HH, 1, kmax), o[nt]);
    __syncthreads();
  }
#pragma unroll
  for (int nt = 0; nt < 4; ++nt)
#pragma unroll
    for (int i = 0; i < 8; ++i){
      float val = o[nt][i] / lrun[i];
      AO[((size_t)bb * TT + qrow0 + i) * HH + hh * HDD + nt * 16 + nc] = (h16)val;
    }
}

// ---------------------------------------------------------------------------
// GLU:  u = a * sigmoid(g)   ([BT,1024] -> [BT,512] f16)
// ---------------------------------------------------------------------------
__global__ void k_glu(const float* __restrict__ ag, h16* __restrict__ u){
  size_t i = (size_t)blockIdx.x * blockDim.x + threadIdx.x;
  if (i < (size_t)BTT * HH){
    size_t row = i >> 9; int c = (int)(i & 511);
    float a = ag[row * 1024 + c];
    float g = ag[row * 1024 + 512 + c];
    u[i] = (h16)(a * sigmf(g));
  }
}

// ---------------------------------------------------------------------------
// Head: [BT,256] @ [3,256]^T + sigmoid + SEIR scaling -> d_out layout
// ---------------------------------------------------------------------------
__global__ void k_head3(const h16* __restrict__ h2, const float* __restrict__ W,
                        const float* __restrict__ b, float* __restrict__ out){
  int row  = blockIdx.x * 8 + (threadIdx.x >> 5);
  int lane = threadIdx.x & 31;
  float l0 = 0, l1 = 0, l2 = 0;
  for (int c = lane; c < 256; c += 32){
    float v = (float)h2[(size_t)row * 256 + c];
    l0 += v * W[0 * 256 + c]; l1 += v * W[1 * 256 + c]; l2 += v * W[2 * 256 + c];
  }
  for (int o = 16; o > 0; o >>= 1){
    l0 += __shfl_xor(l0, o, 32); l1 += __shfl_xor(l1, o, 32); l2 += __shfl_xor(l2, o, 32);
  }
  if (lane == 0){
    out[row]            = sigmf(l0 + b[0]) * 2.5f;
    out[BTT + row]      = sigmf(l1 + b[1]) * (0.5f  - 0.1f)  + 0.1f;
    out[2 * BTT + row]  = sigmf(l2 + b[2]) * (0.35f - 0.03f) + 0.03f;
  }
}

// ---------------------------------------------------------------------------
extern "C" void kernel_launch(void* const* d_in, const int* in_sizes, int n_in,
                              void* d_out, int out_size, void* d_ws, size_t ws_size,
                              hipStream_t stream){
  (void)in_sizes; (void)n_in; (void)out_size; (void)ws_size;
  auto F = [&](int i){ return (const float*)d_in[i]; };
  const float* x_in       = F(0);
  const float* in_norm_w  = F(1);  const float* in_norm_b  = F(2);
  const float* proj_W     = F(3);  const float* proj_b     = F(4);
  const float* proj_nw    = F(5);  const float* proj_nb    = F(6);
  const float* convW[3]   = { F(7),  F(11), F(15) };
  const float* convB[3]   = { F(8),  F(12), F(16) };
  const float* bnG[3]     = { F(9),  F(13), F(17) };
  const float* bnB[3]     = { F(10), F(14), F(18) };
  const float* tgate_W    = F(19); const float* tgate_b    = F(20);
  const float* lW_ih[2]   = { F(21), F(25) };
  const float* lW_hh[2]   = { F(22), F(26) };
  const float* lb_ih[2]   = { F(23), F(27) };
  const float* lb_hh[2]   = { F(24), F(28) };
  const float* q_W = F(29); const float* q_b = F(30);
  const float* k_W = F(31); const float* k_b = F(32);
  const float* v_W = F(33); const float* v_b = F(34);
  const float* o_W = F(35); const float* o_b = F(36);
  const float* rel_pos = F(37);
  const float* attn_nw = F(38); const float* attn_nb = F(39);
  const float* gate_W = F(40);  const float* gate_b = F(41);
  const float* ff_nw = F(42);   const float* ff_nb = F(43);
  const float* ffg_W = F(44);   const float* ffg_b = F(45);
  const float* ffp_W = F(46);   const float* ffp_b = F(47);
  const float* ph1_W = F(48);   const float* ph1_b = F(49);
  const float* ph2_W = F(50);   const float* ph2_b = F(51);
  const float* ph3_W = F(52);   const float* ph3_b = F(53);

  // --- workspace bump allocator ---
  char* base = (char*)d_ws; size_t off = 0;
  auto alloc = [&](size_t bytes)->void*{
    void* p = base + off; off = (off + bytes + 255) & ~(size_t)255; return p;
  };
  // f16 weights
  h16* projW16 = (h16*)alloc((size_t)HH * DD * 2);
  h16* convW16 = (h16*)alloc((size_t)3 * 3 * HH * HH * 2);  // [conv][tap][512*512]
  h16* lIh16[2]; h16* lHh16[2];
  for (int l = 0; l < 2; ++l){
    lIh16[l] = (h16*)alloc((size_t)4 * HH * HH * 2);
    lHh16[l] = (h16*)alloc((size_t)4 * HH * HH * 2);
  }
  h16* qW16 = (h16*)alloc((size_t)HH * HH * 2);
  h16* kW16 = (h16*)alloc((size_t)HH * HH * 2);
  h16* vW16 = (h16*)alloc((size_t)HH * HH * 2);
  h16* oW16 = (h16*)alloc((size_t)HH * HH * 2);
  h16* gW16  = (h16*)alloc((size_t)HH * 2 * HH * 2);
  h16* fgW16 = (h16*)alloc((size_t)2 * HH * HH * 2);
  h16* fpW16 = (h16*)alloc((size_t)HH * HH * 2);
  h16* p1W16 = (h16*)alloc((size_t)HH * HH * 2);
  h16* p2W16 = (h16*)alloc((size_t)(HH/2) * HH * 2);
  // f16 activations
  h16* xln16   = (h16*)alloc((size_t)BTT * DD * 2);
  h16* x16     = (h16*)alloc((size_t)BTT * HH * 2);
  h16* s16a    = (h16*)alloc((size_t)BTT * HH * 2);
  h16* s16b    = (h16*)alloc((size_t)BTT * HH * 2);
  h16* seq16   = (h16*)alloc((size_t)BTT * HH * 2);
  h16* concat16= (h16*)alloc((size_t)BTT * 2 * HH * 2);
  h16* q16     = (h16*)alloc((size_t)BTT * HH * 2);
  h16* k16     = (h16*)alloc((size_t)BTT * HH * 2);
  h16* v16     = (h16*)alloc((size_t)BTT * HH * 2);
  h16* ao16    = (h16*)alloc((size_t)BTT * HH * 2);
  h16* hbuf    = (h16*)alloc((size_t)2 * BB * HH * 2);
  // f32 activations
  float* tmpA  = (float*)alloc((size_t)BTT * HH * 4);       // proj out -> lstm_out
  float* c1    = (float*)alloc((size_t)BTT * HH * 4);
  float* c2    = (float*)alloc((size_t)BTT * HH * 4);
  float* c3    = (float*)alloc((size_t)BTT * HH * 4);
  float* xf32  = (float*)alloc((size_t)BTT * HH * 4);
  float* attn32= (float*)alloc((size_t)BTT * HH * 4);
  float* gate32= (float*)alloc((size_t)BTT * HH * 4);
  float* ag32  = (float*)alloc((size_t)BTT * 2 * HH * 4);

  auto cast = [&](const float* s, h16* d, long n){
    k_cast_h16<<<(unsigned)((n + 255) / 256), 256, 0, stream>>>(s, d, n);
  };
  // --- weight casts (all L2-resident for the whole pass) ---
  cast(proj_W, projW16, (long)HH * DD);
  for (int i = 0; i < 3; ++i)
    k_cast_convw<<<(unsigned)(((long)3 * HH * HH + 255) / 256), 256, 0, stream>>>(
        convW[i], convW16 + (size_t)i * 3 * HH * HH);
  for (int l = 0; l < 2; ++l){
    cast(lW_ih[l], lIh16[l], (long)4 * HH * HH);
    cast(lW_hh[l], lHh16[l], (long)4 * HH * HH);
  }
  cast(q_W, qW16, (long)HH * HH);  cast(k_W, kW16, (long)HH * HH);
  cast(v_W, vW16, (long)HH * HH);  cast(o_W, oW16, (long)HH * HH);
  cast(gate_W, gW16, (long)HH * 2 * HH);
  cast(ffg_W, fgW16, (long)2 * HH * HH);
  cast(ffp_W, fpW16, (long)HH * HH);
  cast(ph1_W, p1W16, (long)HH * HH);
  cast(ph2_W, p2W16, (long)(HH/2) * HH);

  // --- pipeline ---
  k_ln32<<<BTT / 8, 256, 0, stream>>>(x_in, in_norm_w, in_norm_b, xln16);
  k_gemm<1,false><<<dim3(HH/64, BTT/64), 256, 0, stream>>>(
      xln16, projW16, proj_b, nullptr, tmpA, nullptr, BTT, HH, DD, 0);
  k_ln512<<<BTT, 256, 0, stream>>>(tmpA, nullptr, nullptr, proj_nw, proj_nb, xf32, x16);

  int dils[3] = {1, 2, 4};
  float* couts[3] = {c1, c2, c3};
  for (int i = 0; i < 3; ++i)
    k_convgemm<<<dim3(HH/64, BTT/64), 256, 0, stream>>>(
        x16, convW16 + (size_t)i * 3 * HH * HH, convB[i], bnG[i], bnB[i],
        xf32, couts[i], dils[i]);
  k_tmix<<<BTT / 8, 256, 0, stream>>>(c1, c2, c3, tgate_W, tgate_b, xf32, x16);

  // LSTM layers (persistent, one conceptual 16-WG cluster each)
  k_lstm<<<16, 256, 0, stream>>>(x16, lIh16[0], lHh16[0], lb_ih[0], lb_hh[0],
                                 hbuf, s16a, HH, nullptr, 0, nullptr);
  k_lstm<<<16, 256, 0, stream>>>(s16a, lIh16[1], lHh16[1], lb_ih[1], lb_hh[1],
                                 hbuf, concat16, 2 * HH, seq16, HH, tmpA);

  // attention
  k_gemm<0,false><<<dim3(HH/64, BTT/64), 256, 0, stream>>>(
      seq16, qW16, q_b, nullptr, nullptr, q16, BTT, HH, HH, HH);
  k_gemm<0,false><<<dim3(HH/64, BTT/64), 256, 0, stream>>>(
      seq16, kW16, k_b, nullptr, nullptr, k16, BTT, HH, HH, HH);
  k_gemm<0,false><<<dim3(HH/64, BTT/64), 256, 0, stream>>>(
      seq16, vW16, v_b, nullptr, nullptr, v16, BTT, HH, HH, HH);
  k_attn<<<dim3(TT/16, NHH, BB), 32, 0, stream>>>(q16, k16, v16, rel_pos, ao16);
  k_gemm<0,false><<<dim3(HH/64, BTT/64), 256, 0, stream>>>(
      ao16, oW16, o_b, nullptr, attn32, concat16 + HH, BTT, HH, HH, 2 * HH);

  // gated residual + LN
  k_gemm<2,false><<<dim3(HH/64, BTT/64), 256, 0, stream>>>(
      concat16, gW16, gate_b, nullptr, gate32, nullptr, BTT, HH, 2 * HH, 0);
  k_ln512<<<BTT, 256, 0, stream>>>(tmpA, gate32, attn32, attn_nw, attn_nb, xf32, x16);

  // FF GLU block
  k_ln512<<<BTT, 256, 0, stream>>>(xf32, nullptr, nullptr, ff_nw, ff_nb, nullptr, s16a);
  k_gemm<0,false><<<dim3(2*HH/64, BTT/64), 256, 0, stream>>>(
      s16a, fgW16, ffg_b, nullptr, ag32, nullptr, BTT, 2 * HH, HH, 0);
  k_glu<<<(unsigned)(((size_t)BTT * HH + 255) / 256), 256, 0, stream>>>(ag32, s16b);
  k_gemm<0,true><<<dim3(HH/64, BTT/64), 256, 0, stream>>>(
      s16b, fpW16, ffp_b, xf32, nullptr, x16, BTT, HH, HH, HH);

  // param head
  k_gemm<1,false><<<dim3(HH/64, BTT/64), 256, 0, stream>>>(
      x16, p1W16, ph1_b, nullptr, nullptr, s16a, BTT, HH, HH, HH);
  k_gemm<1,false><<<dim3((HH/2)/64, BTT/64), 256, 0, stream>>>(
      s16a, p2W16, ph2_b, nullptr, nullptr, s16b, BTT, HH/2, HH, HH/2);
  k_head3<<<BTT / 8, 256, 0, stream>>>(s16b, ph3_W, ph3_b, (float*)d_out);
}
